// Attention_16767552323667
// MI455X (gfx1250) — matrix operand order repaired and checked by reading, not compiled, not run on hardware
//
#include <hip/hip_runtime.h>
#include <stdint.h>

#define S_      32
#define HQ_     32
#define HKV_    8
#define D_      128
#define BS_     16
#define MAXBLK_ 128
#define G_      4           // HQ / HKV
#define NW_     8           // waves per workgroup
#define SCALE_  0.08838834764831845f   // 128^-0.5

typedef __attribute__((ext_vector_type(16))) __bf16 v16bf;
typedef __attribute__((ext_vector_type(8)))  float  v8f;
typedef __attribute__((ext_vector_type(4)))  float  f4;

union FragAB { v16bf v; uint32_t d[8]; };

// Pack two f32 into a bf16 pair (truncating) with a single v_perm_b32:
// dst[15:0] = lo[31:16], dst[31:16] = hi[31:16].
__device__ __forceinline__ uint32_t packbf(float lo, float hi) {
    union { float f; uint32_t u; } a, b; a.f = lo; b.f = hi;
    return __builtin_amdgcn_perm(b.u, a.u, 0x07060302u);
}
__device__ __forceinline__ unsigned short bfhi(float x) {
    union { float f; uint32_t u; } c; c.f = x;
    return (unsigned short)(c.u >> 16);    // truncating f32 -> bf16
}

__global__ __launch_bounds__(256)
void paged_decode_attn(const float* __restrict__ q,
                       const float* __restrict__ k,
                       const float* __restrict__ v,
                       const float* __restrict__ k_cache,
                       const float* __restrict__ v_cache,
                       const int*   __restrict__ ctx_lens,
                       const int*   __restrict__ btab,
                       float*       __restrict__ out)
{
    const int s    = blockIdx.x / HKV_;
    const int h    = blockIdx.x % HKV_;
    const int tid  = threadIdx.x;
    const int w    = tid >> 5;
    const int lane = tid & 31;
    const int half = lane >> 4;
    const int n    = lane & 15;
    const int ctx  = ctx_lens[s];
    const int last = ctx - 1;
    const int nchunks = (ctx + 31) >> 5;     // 32 tokens (2 cache blocks) per chunk

    __shared__ unsigned short Pbuf[NW_][16 * 34];   // per-wave P transpose bounce (bf16)
    __shared__ float Oacc[16][D_];                  // cross-wave O accumulator
    __shared__ float mstat[NW_][16];
    __shared__ float lstat[NW_][16];

    for (int i = tid; i < 16 * D_; i += 256) ((float*)Oacc)[i] = 0.f;

    // ---- Q A-fragments (pre-scaled, bf16). A layout: M=lane&15; dword r holds
    // K = (r&3)*2 + (r>>2)*16 + 8*half, K+1. Rows M>=4 zero-padded. ----
    FragAB aq[4];
    {
        const float* qrow = q + ((size_t)s * HQ_ + h * G_ + n) * D_;
        #pragma unroll
        for (int st = 0; st < 4; ++st) {
            #pragma unroll
            for (int r = 0; r < 8; ++r) {
                const int kd = st * 32 + (r & 3) * 2 + (r >> 2) * 16 + half * 8;
                float f0 = 0.f, f1 = 0.f;
                if (n < G_) { f0 = qrow[kd] * SCALE_; f1 = qrow[kd + 1] * SCALE_; }
                aq[st].d[r] = packbf(f0, f1);
            }
        }
    }

    const v8f vzero = {0.f,0.f,0.f,0.f,0.f,0.f,0.f,0.f};
    float mrun[8], lrun[8];
    v8f oc[8];
    #pragma unroll
    for (int j = 0; j < 8; ++j) { mrun[j] = -1e30f; lrun[j] = 0.f; }
    #pragma unroll
    for (int dt = 0; dt < 8; ++dt) oc[dt] = vzero;

    const float* knew = k + ((size_t)s * HKV_ + h) * D_;
    const float* vnew = v + ((size_t)s * HKV_ + h) * D_;

    for (int c = w; c < nchunks; c += NW_) {
        const int base = c * 32;
        const int blk0 = btab[s * MAXBLK_ + (base >> 4)];
        const int blk1 = btab[s * MAXBLK_ + (base >> 4) + 1];

        // prefetch next chunk owned by this wave
        {
            const int cn = c + NW_;
            if (cn < nchunks) {
                const int b = btab[s * MAXBLK_ + cn * 2];
                const size_t off = (((size_t)(b * BS_ + n) * HKV_) + h) * D_ + half * 64;
                __builtin_prefetch(k_cache + off, 0, 1);
                __builtin_prefetch(v_cache + off, 0, 1);
            }
        }

        // ---- S = Q @ K^T : two 16-token N tiles, contraction over d=128 ----
        v8f sc[2];
        #pragma unroll
        for (int nt = 0; nt < 2; ++nt) {
            sc[nt] = vzero;
            const int tik = nt * 16 + n;                 // token within chunk (this lane's column)
            const int l   = base + tik;                  // logical position
            const int row = (nt ? blk1 : blk0) * BS_ + n;
            const float* kp = (l == last) ? knew
                            : (k_cache + (((size_t)row * HKV_) + h) * D_);
            #pragma unroll
            for (int st = 0; st < 4; ++st) {
                const float* p16 = kp + st * 32 + half * 8;    // 8 fp32 at +0 and 8 at +16
                f4 x0 = *(const f4*)(p16 + 0);
                f4 x1 = *(const f4*)(p16 + 4);
                f4 x2 = *(const f4*)(p16 + 16);
                f4 x3 = *(const f4*)(p16 + 20);
                FragAB bk;   // B layout: dword r = K rows (r&3)*2 + (r>>2)*16 + 8*half, +1, N=lane&15
                bk.d[0] = packbf(x0.x, x0.y); bk.d[1] = packbf(x0.z, x0.w);
                bk.d[2] = packbf(x1.x, x1.y); bk.d[3] = packbf(x1.z, x1.w);
                bk.d[4] = packbf(x2.x, x2.y); bk.d[5] = packbf(x2.z, x2.w);
                bk.d[6] = packbf(x3.x, x3.y); bk.d[7] = packbf(x3.z, x3.w);
                sc[nt] = __builtin_amdgcn_wmma_f32_16x16x32_bf16(
                    false, aq[st].v, false, bk.v, (short)0, sc[nt], false, false);
            }
        }

        // ---- online softmax (rows j+8*half live in this lane's 8 VGPRs) ----
        const bool ok0 = (base + n)      < ctx;
        const bool ok1 = (base + 16 + n) < ctx;
        float s0[8], s1[8], rmax[8];
        #pragma unroll
        for (int j = 0; j < 8; ++j) {
            s0[j] = ok0 ? sc[0][j] : -1e30f;
            s1[j] = ok1 ? sc[1][j] : -1e30f;
            rmax[j] = fmaxf(s0[j], s1[j]);
        }
        #pragma unroll
        for (int m = 1; m <= 8; m <<= 1)
            #pragma unroll
            for (int j = 0; j < 8; ++j)
                rmax[j] = fmaxf(rmax[j], __shfl_xor(rmax[j], m, 32));
        v8f alphav;
        float rsum[8];
        #pragma unroll
        for (int j = 0; j < 8; ++j) {
            const float mnew = fmaxf(mrun[j], rmax[j]);
            alphav[j] = __expf(mrun[j] - mnew);
            mrun[j]  = mnew;
            s0[j] = __expf(s0[j] - mnew);
            s1[j] = __expf(s1[j] - mnew);
            rsum[j] = s0[j] + s1[j];
        }
        #pragma unroll
        for (int m = 1; m <= 8; m <<= 1)
            #pragma unroll
            for (int j = 0; j < 8; ++j)
                rsum[j] += __shfl_xor(rsum[j], m, 32);
        #pragma unroll
        for (int j = 0; j < 8; ++j) lrun[j] = lrun[j] * alphav[j] + rsum[j];
        // vector rescale of O accumulators (packed/dual-issue f32 muls)
        #pragma unroll
        for (int dt = 0; dt < 8; ++dt) oc[dt] = oc[dt] * alphav;

        // ---- P: C-layout -> A-layout via per-wave LDS bounce (bf16) ----
        unsigned short* pb = &Pbuf[w][0];
        #pragma unroll
        for (int j = 0; j < 8; ++j) {
            const int rowm = j + half * 8;
            pb[rowm * 34 + n]      = bfhi(s0[j]);
            pb[rowm * 34 + 16 + n] = bfhi(s1[j]);
        }
        asm volatile("" ::: "memory");   // same-wave DS ops are in-order; block compiler reorder
        FragAB ap;
        #pragma unroll
        for (int r = 0; r < 8; ++r) {
            const int kd = half * 16 + r * 2;  // even -> 4B aligned
            ap.d[r] = *(const uint32_t*)&pb[n * 34 + kd];
        }

        // ---- O += P @ V : B = V[K=token 32][N=d 16-col tile] ----
        // Wave-uniform split: only the single chunk containing the decoded token
        // needs per-token pointer overrides; everything else uses compact
        // 32-bit offsets off a uniform base (saddr+voffset addressing).
        const bool haslast = ((unsigned)(last - base)) < 32u;   // uniform per wave
        if (!haslast) {
            uint32_t voff[16];                         // element offsets into v_cache
            #pragma unroll
            for (int j2 = 0; j2 < 16; ++j2) {
                const int tik = half * 16 + j2;
                const int row = (tik < 16 ? blk0 : blk1) * BS_ + (tik & 15);
                voff[j2] = ((uint32_t)row * HKV_ + h) * D_ + n;
            }
            #pragma unroll
            for (int dt = 0; dt < 8; ++dt) {
                float va[16];
                #pragma unroll
                for (int j2 = 0; j2 < 16; ++j2)        // batch 16 loads, single wait
                    va[j2] = v_cache[voff[j2] + dt * 16];
                FragAB bv;
                #pragma unroll
                for (int r = 0; r < 8; ++r)
                    bv.d[r] = packbf(va[2 * r], va[2 * r + 1]);
                oc[dt] = __builtin_amdgcn_wmma_f32_16x16x32_bf16(
                    false, ap.v, false, bv.v, (short)0, oc[dt], false, false);
            }
        } else {
            const float* vp[16];
            #pragma unroll
            for (int j2 = 0; j2 < 16; ++j2) {
                const int tik = half * 16 + j2;
                const int l   = base + tik;
                const int row = (tik < 16 ? blk0 : blk1) * BS_ + (tik & 15);
                vp[j2] = (l == last) ? vnew
                       : (v_cache + (((size_t)row * HKV_) + h) * D_);
            }
            #pragma unroll
            for (int dt = 0; dt < 8; ++dt) {
                float va[16];
                #pragma unroll
                for (int j2 = 0; j2 < 16; ++j2)
                    va[j2] = vp[j2][dt * 16 + n];
                FragAB bv;
                #pragma unroll
                for (int r = 0; r < 8; ++r)
                    bv.d[r] = packbf(va[2 * r], va[2 * r + 1]);
                oc[dt] = __builtin_amdgcn_wmma_f32_16x16x32_bf16(
                    false, ap.v, false, bv.v, (short)0, oc[dt], false, false);
            }
        }
    }

    // ---- cross-wave flash-decoding combine ----
    if (lane == 0 || lane == 16) {
        #pragma unroll
        for (int j = 0; j < 8; ++j) {
            mstat[w][j + half * 8] = mrun[j];
            lstat[w][j + half * 8] = lrun[j];
        }
    }
    __syncthreads();

    v8f amv;
    #pragma unroll
    for (int j = 0; j < 8; ++j) {
        float mg = -1e30f;
        for (int w2 = 0; w2 < NW_; ++w2) mg = fmaxf(mg, mstat[w2][j + half * 8]);
        amv[j] = __expf(mrun[j] - mg);
    }
    #pragma unroll
    for (int dt = 0; dt < 8; ++dt) {
        const v8f sc8 = oc[dt] * amv;
        #pragma unroll
        for (int j = 0; j < 8; ++j)
            atomicAdd(&Oacc[j + half * 8][dt * 16 + n], sc8[j]);
    }
    __syncthreads();

    if (tid < D_) {
        #pragma unroll
        for (int g = 0; g < G_; ++g) {
            float mg = -1e30f;
            for (int w2 = 0; w2 < NW_; ++w2) mg = fmaxf(mg, mstat[w2][g]);
            float Lg = 0.f;
            for (int w2 = 0; w2 < NW_; ++w2)
                Lg += lstat[w2][g] * __expf(mstat[w2][g] - mg);
            out[((size_t)s * HQ_ + h * G_ + g) * D_ + tid] = Oacc[g][tid] / Lg;
        }
    }
}

extern "C" void kernel_launch(void* const* d_in, const int* in_sizes, int n_in,
                              void* d_out, int out_size, void* d_ws, size_t ws_size,
                              hipStream_t stream) {
    const float* q  = (const float*)d_in[0];
    const float* k  = (const float*)d_in[1];
    const float* v  = (const float*)d_in[2];
    const float* kc = (const float*)d_in[3];
    const float* vc = (const float*)d_in[4];
    const int*   cl = (const int*)d_in[5];
    const int*   bt = (const int*)d_in[6];
    // d_in[7] slot_mapping: not needed (derived from ctx_lens + block_tables;
    // the KV scatter is emulated by overriding token ctx-1 with k/v, so the
    // const input caches are never mutated).
    (void)in_sizes; (void)n_in; (void)out_size; (void)d_ws; (void)ws_size;
    paged_decode_attn<<<dim3(S_ * HKV_), dim3(256), 0, stream>>>(
        q, k, v, kc, vc, cl, bt, (float*)d_out);
}
